// SpatialTransformer_67662914781905
// MI455X (gfx1250) — matrix operand order repaired and checked by reading, not compiled, not run on hardware
//
#include <hip/hip_runtime.h>
#include <hip/hip_bf16.h>

// ---------------- CDNA5 WMMA types ----------------
typedef __attribute__((ext_vector_type(16))) __bf16 v16bf;
typedef __attribute__((ext_vector_type(8)))  __bf16 v8bf;
typedef __attribute__((ext_vector_type(8)))  float  v8f;

#define C_DIM  640
#define TOKENS 8192
#define NHEAD  8
#define DHEAD  80

// CDNA5 async global->LDS path (ASYNCcnt), with safe fallback
#if defined(__HIP_DEVICE_COMPILE__) && \
    __has_builtin(__builtin_amdgcn_global_load_async_to_lds_b128) && \
    __has_builtin(__builtin_amdgcn_s_wait_asynccnt)
#define USE_ASYNC_LDS 1
#else
#define USE_ASYNC_LDS 0
#endif

typedef __attribute__((__vector_size__(16))) int vi4;

__device__ __forceinline__ void copy_b128_g2l(const unsigned short* src,
                                              unsigned short* dst)
{
#if USE_ASYNC_LDS
  __builtin_amdgcn_global_load_async_to_lds_b128(
      (__attribute__((address_space(1))) vi4*)src,
      (__attribute__((address_space(3))) vi4*)dst, 0, 0);
#else
  *(uint4*)dst = *(const uint4*)src;
#endif
}

#if USE_ASYNC_LDS
#define ASYNC_WAIT(n) __builtin_amdgcn_s_wait_asynccnt(n)
#else
#define ASYNC_WAIT(n)
#endif

// round-to-nearest-even f32 -> bf16 bits
__device__ __forceinline__ unsigned short f2bf(float f){
  unsigned int u = __float_as_uint(f);
  u += 0x7fffu + ((u >> 16) & 1u);
  return (unsigned short)(u >> 16);
}

__device__ __forceinline__ v16bf cat16(v8bf lo, v8bf hi){
  return __builtin_shufflevector(lo, hi, 0,1,2,3,4,5,6,7,8,9,10,11,12,13,14,15);
}

// =====================================================================
// elementwise f32 -> bf16
// =====================================================================
__global__ __launch_bounds__(256) void f32_to_bf16_kernel(
    const float* __restrict__ in, unsigned short* __restrict__ out, int n)
{
  int i = blockIdx.x * 256 + threadIdx.x;
  if (i < n) out[i] = f2bf(in[i]);
}

// =====================================================================
// convert + transpose: in f32 [K][N] -> out bf16 [N][K] (weights, once/launch)
// =====================================================================
__global__ __launch_bounds__(256) void cvt_transpose_kernel(
    const float* __restrict__ in, unsigned short* __restrict__ out, int K, int N)
{
  __shared__ float tile[32][33];
  int k0 = blockIdx.y * 32, n0 = blockIdx.x * 32;
  for (int i = threadIdx.y; i < 32; i += 8) {
    int k = k0 + i, n = n0 + threadIdx.x;
    tile[i][threadIdx.x] = (k < K && n < N) ? in[(size_t)k * N + n] : 0.0f;
  }
  __syncthreads();
  for (int i = threadIdx.y; i < 32; i += 8) {
    int n = n0 + i, k = k0 + threadIdx.x;
    if (n < N && k < K) out[(size_t)n * K + k] = f2bf(tile[threadIdx.x][i]);
  }
}

// =====================================================================
// GroupNorm: stats per (batch, group) over 32*32 positions x 20 channels
// =====================================================================
__global__ __launch_bounds__(256) void gn_stats_kernel(
    const float* __restrict__ x, float* __restrict__ stats)
{
  __shared__ float s1[256], s2[256];
  const int bg = blockIdx.x;           // b*32 + g
  const int b = bg >> 5, g = bg & 31;
  const int tid = threadIdx.x;
  float sum = 0.f, ss = 0.f;
  for (int i = tid; i < 20480; i += 256) {
    int p = i / 20, cc = i % 20;
    float v = x[((size_t)b * 1024 + p) * C_DIM + g * 20 + cc];
    sum += v; ss += v * v;
  }
  s1[tid] = sum; s2[tid] = ss;
  __syncthreads();
  for (int st = 128; st; st >>= 1) {
    if (tid < st) { s1[tid] += s1[tid + st]; s2[tid] += s2[tid + st]; }
    __syncthreads();
  }
  if (tid == 0) {
    float mean = s1[0] * (1.0f / 20480.0f);
    float var  = s2[0] * (1.0f / 20480.0f) - mean * mean;
    stats[bg * 2]     = mean;
    stats[bg * 2 + 1] = rsqrtf(var + 1e-5f);
  }
}

__global__ __launch_bounds__(256) void gn_apply_kernel(
    const float* __restrict__ x, const float* __restrict__ stats,
    const float* __restrict__ gamma, const float* __restrict__ beta,
    unsigned short* __restrict__ out)
{
  int i = blockIdx.x * 256 + threadIdx.x;      // over TOKENS*C_DIM
  int c = i % C_DIM;
  int tok = i / C_DIM;
  int b = tok >> 10;
  int g = c / 20;
  float mean = stats[(b * 32 + g) * 2];
  float rstd = stats[(b * 32 + g) * 2 + 1];
  out[i] = f2bf((x[i] - mean) * rstd * gamma[c] + beta[c]);
}

// =====================================================================
// LayerNorm over 640 channels, one wave per token, bf16 out
// =====================================================================
__global__ __launch_bounds__(256) void layernorm_kernel(
    const float* __restrict__ x, const float* __restrict__ g,
    const float* __restrict__ bta, unsigned short* __restrict__ out, int ntok)
{
  int wave = threadIdx.x >> 5, lane = threadIdx.x & 31;
  int tok = blockIdx.x * 8 + wave;
  if (tok >= ntok) return;
  const float* xp = x + (size_t)tok * C_DIM;
  float sum = 0.f, ss = 0.f;
  #pragma unroll
  for (int i = 0; i < 20; ++i) {
    float v = xp[lane * 20 + i];
    sum += v; ss += v * v;
  }
  #pragma unroll
  for (int off = 16; off; off >>= 1) {
    sum += __shfl_xor(sum, off, 32);
    ss  += __shfl_xor(ss,  off, 32);
  }
  float mean = sum * (1.0f / 640.0f);
  float var  = ss  * (1.0f / 640.0f) - mean * mean;
  float rstd = rsqrtf(var + 1e-5f);
  unsigned short* op = out + (size_t)tok * C_DIM;
  #pragma unroll
  for (int i = 0; i < 20; ++i) {
    int c = lane * 20 + i;
    op[c] = f2bf((xp[c] - mean) * rstd * g[c] + bta[c]);
  }
}

// =====================================================================
// bf16 WMMA GEMM: out[MxN] = A[MxK](bf16,row-major) * W (bf16, PRE-TRANSPOSED
// [N][K]) (+bias)(+res). Block tile 128x64, 8 waves (4Mx2N), wave 32x32 via
// 2x2 wmma. Double-buffered LDS; async global->LDS when available.
// =====================================================================
__global__ __launch_bounds__(256) void gemm_kernel(
    const unsigned short* __restrict__ A, int lda,
    const unsigned short* __restrict__ Bt,      // [N][K]
    const float* __restrict__ bias,
    const float* res, float* outF, unsigned short* outB, int ldo,
    int M, int N, int K)
{
  __shared__ unsigned short sA[2 * 128 * 40];
  __shared__ unsigned short sB[2 * 64 * 40];
  const int tid  = threadIdx.x;
  const int wave = tid >> 5, lane = tid & 31;
  const int hl = (lane >= 16) ? 1 : 0;
  const int rA = lane & 15;
  const int wm = (wave & 3) * 32;
  const int wn = (wave >> 2) * 32;
  const int m0 = blockIdx.y * 128;
  const int n0 = blockIdx.x * 64;
  (void)N;

  v8f acc[2][2] = {};
  const int nk = K >> 5;

  auto compute = [&](int buf) {
    const unsigned short* bA = sA + buf * (128 * 40);
    const unsigned short* bB = sB + buf * (64 * 40);
    v16bf af[2], bf[2];
    #pragma unroll
    for (int i = 0; i < 2; ++i) {
      const unsigned short* pa = bA + (wm + i * 16 + rA) * 40 + (hl ? 8 : 0);
      af[i] = cat16(*(const v8bf*)pa, *(const v8bf*)(pa + 16));
    }
    #pragma unroll
    for (int j = 0; j < 2; ++j) {
      const unsigned short* pb = bB + (wn + j * 16 + rA) * 40 + (hl ? 8 : 0);
      bf[j] = cat16(*(const v8bf*)pb, *(const v8bf*)(pb + 16));
    }
    #pragma unroll
    for (int i = 0; i < 2; ++i)
      #pragma unroll
      for (int j = 0; j < 2; ++j)
        acc[i][j] = __builtin_amdgcn_wmma_f32_16x16x32_bf16(
            false, af[i], false, bf[j], (short)0, acc[i][j], false, false);
  };

  if (m0 + 128 <= M) {
    // ---------------- full-tile pipelined path ----------------
    auto stage = [&](int buf, int k0) {
      #pragma unroll
      for (int it = 0; it < 2; ++it) {
        int c = tid + it * 256;
        int r = c >> 2, kc = (c & 3) * 8;
        copy_b128_g2l(A + (size_t)(m0 + r) * lda + k0 + kc,
                      sA + buf * (128 * 40) + r * 40 + kc);
      }
      {
        int r = tid >> 2, kc = (tid & 3) * 8;
        copy_b128_g2l(Bt + (size_t)(n0 + r) * K + k0 + kc,
                      sB + buf * (64 * 40) + r * 40 + kc);
      }
    };
    stage(0, 0);
    for (int ks = 0; ks < nk; ++ks) {
      const int buf = ks & 1;
      if (ks + 1 < nk) {
        stage(buf ^ 1, (ks + 1) * 32);
        ASYNC_WAIT(3);          // previous group done (in-order completion)
      } else {
        ASYNC_WAIT(0);
      }
      __syncthreads();
      compute(buf);
      __syncthreads();
    }
  } else {
    // ---------------- guarded edge path (single buffer) ----------------
    for (int ks = 0; ks < nk; ++ks) {
      const int k0 = ks * 32;
      __syncthreads();
      #pragma unroll
      for (int it = 0; it < 2; ++it) {
        int c = tid + it * 256;
        int r = c >> 2, kc = (c & 3) * 8;
        uint4 v = make_uint4(0, 0, 0, 0);
        if (m0 + r < M) v = *(const uint4*)(A + (size_t)(m0 + r) * lda + k0 + kc);
        *(uint4*)(sA + r * 40 + kc) = v;
      }
      {
        int r = tid >> 2, kc = (tid & 3) * 8;
        *(uint4*)(sB + r * 40 + kc) =
            *(const uint4*)(Bt + (size_t)(n0 + r) * K + k0 + kc);
      }
      __syncthreads();
      compute(0);
    }
  }

  // epilogue: C/D layout -> lane l, VGPR r holds (M = r + 8*hl, N = l&15)
  #pragma unroll
  for (int i = 0; i < 2; ++i) {
    #pragma unroll
    for (int j = 0; j < 2; ++j) {
      int col = n0 + wn + j * 16 + rA;
      float bv = bias ? bias[col] : 0.0f;
      #pragma unroll
      for (int r = 0; r < 8; ++r) {
        int row = m0 + wm + i * 16 + r + 8 * hl;
        if (row < M) {
          float v = acc[i][j][r] + bv;
          if (res)  v += res[(size_t)row * ldo + col];
          if (outF) outF[(size_t)row * ldo + col] = v;
          if (outB) outB[(size_t)row * ldo + col] = f2bf(v);
        }
      }
    }
  }
}

// =====================================================================
// GEGLU-fused GEMM: y = A*Wg + b; xx = y[:, n], gate = y[:, n+Nh]
// out = xx * 0.5 * gate * (1 + tanh(0.79788456*gate*(1+0.044715*gate^2)))
// Wg pre-transposed [5120][K]; gate rows at n+Nh. Full tiles only (M=8192).
// =====================================================================
__global__ __launch_bounds__(256) void gemm_geglu_kernel(
    const unsigned short* __restrict__ A, int lda,
    const unsigned short* __restrict__ Bt,   // [2*Nh][K]
    const float* __restrict__ bias,          // 2*Nh
    unsigned short* __restrict__ outB,       // M x Nh
    int M, int Nh, int K)
{
  __shared__ unsigned short sA[2 * 128 * 40];
  __shared__ unsigned short sX[2 * 64 * 40];
  __shared__ unsigned short sG[2 * 64 * 40];
  const int tid  = threadIdx.x;
  const int wave = tid >> 5, lane = tid & 31;
  const int hl = (lane >= 16) ? 1 : 0;
  const int rA = lane & 15;
  const int wm = (wave & 3) * 32;
  const int wn = (wave >> 2) * 32;
  const int m0 = blockIdx.y * 128;
  const int n0 = blockIdx.x * 64;

  v8f accx[2][2] = {};
  v8f accg[2][2] = {};
  const int nk = K >> 5;

  auto stage = [&](int buf, int k0) {
    #pragma unroll
    for (int it = 0; it < 2; ++it) {
      int c = tid + it * 256;
      int r = c >> 2, kc = (c & 3) * 8;
      copy_b128_g2l(A + (size_t)(m0 + r) * lda + k0 + kc,
                    sA + buf * (128 * 40) + r * 40 + kc);
    }
    int r = tid >> 2, kc = (tid & 3) * 8;
    copy_b128_g2l(Bt + (size_t)(n0 + r) * K + k0 + kc,
                  sX + buf * (64 * 40) + r * 40 + kc);
    copy_b128_g2l(Bt + (size_t)(n0 + r + Nh) * K + k0 + kc,
                  sG + buf * (64 * 40) + r * 40 + kc);
  };

  stage(0, 0);
  for (int ks = 0; ks < nk; ++ks) {
    const int buf = ks & 1;
    if (ks + 1 < nk) {
      stage(buf ^ 1, (ks + 1) * 32);
      ASYNC_WAIT(4);
    } else {
      ASYNC_WAIT(0);
    }
    __syncthreads();

    const unsigned short* bA = sA + buf * (128 * 40);
    const unsigned short* bX = sX + buf * (64 * 40);
    const unsigned short* bG = sG + buf * (64 * 40);
    v16bf af[2], bx[2], bg[2];
    #pragma unroll
    for (int i = 0; i < 2; ++i) {
      const unsigned short* pa = bA + (wm + i * 16 + rA) * 40 + (hl ? 8 : 0);
      af[i] = cat16(*(const v8bf*)pa, *(const v8bf*)(pa + 16));
    }
    #pragma unroll
    for (int j = 0; j < 2; ++j) {
      const unsigned short* px = bX + (wn + j * 16 + rA) * 40 + (hl ? 8 : 0);
      const unsigned short* pg = bG + (wn + j * 16 + rA) * 40 + (hl ? 8 : 0);
      bx[j] = cat16(*(const v8bf*)px, *(const v8bf*)(px + 16));
      bg[j] = cat16(*(const v8bf*)pg, *(const v8bf*)(pg + 16));
    }
    #pragma unroll
    for (int i = 0; i < 2; ++i)
      #pragma unroll
      for (int j = 0; j < 2; ++j) {
        accx[i][j] = __builtin_amdgcn_wmma_f32_16x16x32_bf16(
            false, af[i], false, bx[j], (short)0, accx[i][j], false, false);
        accg[i][j] = __builtin_amdgcn_wmma_f32_16x16x32_bf16(
            false, af[i], false, bg[j], (short)0, accg[i][j], false, false);
      }
    __syncthreads();
  }

  #pragma unroll
  for (int i = 0; i < 2; ++i) {
    #pragma unroll
    for (int j = 0; j < 2; ++j) {
      int col = n0 + wn + j * 16 + rA;
      float bvx = bias[col];
      float bvg = bias[col + Nh];
      #pragma unroll
      for (int r = 0; r < 8; ++r) {
        int row = m0 + wm + i * 16 + r + 8 * hl;
        if (row < M) {
          float xx = accx[i][j][r] + bvx;
          float gg = accg[i][j][r] + bvg;
          float th = tanhf(0.7978845608f * gg * (1.0f + 0.044715f * gg * gg));
          float hv = xx * 0.5f * gg * (1.0f + th);
          outB[(size_t)row * Nh + col] = f2bf(hv);
        }
      }
    }
  }
}

// =====================================================================
// Flash attention: grid (qblock, head, batch), 8 waves/block share (b,h),
// each wave owns a 16-row Q tile. K/V staged cooperatively per 32-key step
// (K tile async when the step is full). Online softmax, f32 accum.
// =====================================================================
__global__ __launch_bounds__(256) void attn_kernel(
    const unsigned short* __restrict__ Q,
    const unsigned short* __restrict__ Kb,
    const unsigned short* __restrict__ Vb,
    unsigned short* __restrict__ Out,
    int seqK, int strideQb, int strideKVb, float scale)
{
  __shared__ unsigned short sK[32 * 104];     // [key][d] d padded 80->96
  __shared__ unsigned short sV[80 * 40];      // [d][key] transposed
  __shared__ unsigned short sS[8 * 16 * 40];  // per-wave P tile [16][32]

  const int tid  = threadIdx.x;
  const int wave = tid >> 5, lane = tid & 31;
  const int hl = (lane >= 16) ? 1 : 0;
  const int ln = lane & 15;
  const int b = blockIdx.z, h = blockIdx.y;
  const int qrow0 = blockIdx.x * 128 + wave * 16;

  // zero d-pad (80..95) of sK once
  if (tid < 64) {
    uint4 z = make_uint4(0, 0, 0, 0);
    *(uint4*)(sK + (tid >> 1) * 104 + 80 + (tid & 1) * 8) = z;
  }

  // Q fragments (16 x 96 A-layout), zeros in d = 80..95
  v16bf qf[3];
  {
    const unsigned short* qp =
        Q + (size_t)b * strideQb + (size_t)(qrow0 + ln) * C_DIM + h * DHEAD;
    v8bf z8;
    #pragma unroll
    for (int i = 0; i < 8; ++i) z8[i] = (__bf16)0.0f;
    #pragma unroll
    for (int c = 0; c < 3; ++c) {
      int dlo = c * 32 + (hl ? 8 : 0);
      int dhi = dlo + 16;
      v8bf lo = *(const v8bf*)(qp + dlo);
      v8bf hi = (dhi < DHEAD) ? *(const v8bf*)(qp + dhi) : z8;
      qf[c] = cat16(lo, hi);
    }
  }

  float mrow[8], lrow[8];
  #pragma unroll
  for (int r = 0; r < 8; ++r) { mrow[r] = -1e30f; lrow[r] = 0.0f; }
  v8f o[5] = {};

  const int nsteps = (seqK + 31) >> 5;
  for (int ks = 0; ks < nsteps; ++ks) {
    const int k0 = ks * 32;
    const bool full = (k0 + 32 <= seqK);
    __syncthreads();
    // stage K tile [32 keys][d<80]
    if (full) {
      for (int c = tid; c < 320; c += 256) {
        int key = c / 10, dc = (c % 10) * 8;
        copy_b128_g2l(Kb + (size_t)b * strideKVb + (size_t)(k0 + key) * C_DIM +
                          h * DHEAD + dc,
                      sK + key * 104 + dc);
      }
    } else {
      for (int c = tid; c < 320; c += 256) {
        int key = c / 10, dc = (c % 10) * 8;
        uint4 v = make_uint4(0, 0, 0, 0);
        if (k0 + key < seqK)
          v = *(const uint4*)(Kb + (size_t)b * strideKVb +
                              (size_t)(k0 + key) * C_DIM + h * DHEAD + dc);
        *(uint4*)(sK + key * 104 + dc) = v;
      }
    }
    // stage V tile transposed
    for (int c = tid; c < 320; c += 256) {
      int key = c / 10, dc = (c % 10) * 8;
      unsigned short tmp[8] = {0, 0, 0, 0, 0, 0, 0, 0};
      if (k0 + key < seqK) {
        const unsigned short* src = Vb + (size_t)b * strideKVb +
                                    (size_t)(k0 + key) * C_DIM + h * DHEAD + dc;
        #pragma unroll
        for (int i = 0; i < 8; ++i) tmp[i] = src[i];
      }
      #pragma unroll
      for (int i = 0; i < 8; ++i) sV[(dc + i) * 40 + key] = tmp[i];
    }
    ASYNC_WAIT(0);
    __syncthreads();

    // S = Q K^T  (two 16-key N tiles, three K=32 chunks over d)
    v8f s[2] = {};
    #pragma unroll
    for (int f = 0; f < 2; ++f) {
      #pragma unroll
      for (int c = 0; c < 3; ++c) {
        const unsigned short* pb =
            sK + (f * 16 + ln) * 104 + c * 32 + (hl ? 8 : 0);
        v16bf kb = cat16(*(const v8bf*)pb, *(const v8bf*)(pb + 16));
        s[f] = __builtin_amdgcn_wmma_f32_16x16x32_bf16(
            false, qf[c], false, kb, (short)0, s[f], false, false);
      }
    }
    const bool v0 = (k0 + ln) < seqK;
    const bool v1 = (k0 + 16 + ln) < seqK;
    #pragma unroll
    for (int r = 0; r < 8; ++r) {
      s[0][r] = v0 ? s[0][r] * scale : -1e30f;
      s[1][r] = v1 ? s[1][r] * scale : -1e30f;
    }
    // online softmax per row (rows live in 16-lane halves)
    float alpha[8];
    #pragma unroll
    for (int r = 0; r < 8; ++r) {
      float mx = fmaxf(s[0][r], s[1][r]);
      mx = fmaxf(mx, __shfl_xor(mx, 1, 16));
      mx = fmaxf(mx, __shfl_xor(mx, 2, 16));
      mx = fmaxf(mx, __shfl_xor(mx, 4, 16));
      mx = fmaxf(mx, __shfl_xor(mx, 8, 16));
      float nm = fmaxf(mrow[r], mx);
      alpha[r] = __expf(mrow[r] - nm);
      mrow[r] = nm;
      float p0 = v0 ? __expf(s[0][r] - nm) : 0.0f;
      float p1 = v1 ? __expf(s[1][r] - nm) : 0.0f;
      s[0][r] = p0; s[1][r] = p1;
      float ps = p0 + p1;
      ps += __shfl_xor(ps, 1, 16);
      ps += __shfl_xor(ps, 2, 16);
      ps += __shfl_xor(ps, 4, 16);
      ps += __shfl_xor(ps, 8, 16);
      lrow[r] = alpha[r] * lrow[r] + ps;
    }
    // P -> LDS (bf16), then re-read as A-fragment
    unsigned short* sw = sS + wave * 16 * 40;
    #pragma unroll
    for (int r = 0; r < 8; ++r) {
      int row = r + 8 * hl;
      sw[row * 40 + ln]      = f2bf(s[0][r]);
      sw[row * 40 + 16 + ln] = f2bf(s[1][r]);
    }
    __syncthreads();
    v16bf pf;
    {
      const unsigned short* pp = sw + ln * 40 + (hl ? 8 : 0);
      pf = cat16(*(const v8bf*)pp, *(const v8bf*)(pp + 16));
    }
    // O = O*alpha + P V
    #pragma unroll
    for (int f = 0; f < 5; ++f) {
      #pragma unroll
      for (int r = 0; r < 8; ++r) o[f][r] *= alpha[r];
      const unsigned short* pv = sV + (f * 16 + ln) * 40 + (hl ? 8 : 0);
      v16bf vb = cat16(*(const v8bf*)pv, *(const v8bf*)(pv + 16));
      o[f] = __builtin_amdgcn_wmma_f32_16x16x32_bf16(
          false, pf, false, vb, (short)0, o[f], false, false);
    }
  }

  // normalize + store (bf16)
  unsigned short* op = Out + (size_t)b * strideQb + h * DHEAD;
  #pragma unroll
  for (int f = 0; f < 5; ++f) {
    #pragma unroll
    for (int r = 0; r < 8; ++r) {
      int row = qrow0 + r + 8 * hl;
      op[(size_t)row * C_DIM + f * 16 + ln] = f2bf(o[f][r] / lrow[r]);
    }
  }
}

// =====================================================================
// host orchestration
// =====================================================================
extern "C" void kernel_launch(void* const* d_in, const int* in_sizes, int n_in,
                              void* d_out, int out_size, void* d_ws, size_t ws_size,
                              hipStream_t stream)
{
  (void)in_sizes; (void)n_in; (void)out_size; (void)ws_size;
  const float* inputs  = (const float*)d_in[0];
  const float* context = (const float*)d_in[1];
  const float* gn_g    = (const float*)d_in[2];
  const float* gn_b    = (const float*)d_in[3];
  const float* w_proj1 = (const float*)d_in[4];
  const float* b_proj1 = (const float*)d_in[5];
  const float* ln1_g   = (const float*)d_in[6];
  const float* ln1_b   = (const float*)d_in[7];
  const float* wq1     = (const float*)d_in[8];
  const float* wk1     = (const float*)d_in[9];
  const float* wv1     = (const float*)d_in[10];
  const float* wo1     = (const float*)d_in[11];
  const float* bo1     = (const float*)d_in[12];
  const float* ln2_g   = (const float*)d_in[13];
  const float* ln2_b   = (const float*)d_in[14];
  const float* wq2     = (const float*)d_in[15];
  const float* wk2     = (const float*)d_in[16];
  const float* wv2     = (const float*)d_in[17];
  const float* wo2     = (const float*)d_in[18];
  const float* bo2     = (const float*)d_in[19];
  const float* ln3_g   = (const float*)d_in[20];
  const float* ln3_b   = (const float*)d_in[21];
  const float* w_geglu = (const float*)d_in[22];
  const float* b_geglu = (const float*)d_in[23];
  const float* w_ffout = (const float*)d_in[24];
  const float* b_ffout = (const float*)d_in[25];
  const float* w_proj2 = (const float*)d_in[26];
  const float* b_proj2 = (const float*)d_in[27];

  // ---- workspace layout (~103 MB total) ----
  char* ws = (char*)d_ws;
  size_t off = 0;
  auto alloc = [&](size_t bytes) -> void* {
    void* p = ws + off;
    off = (off + bytes + 255) & ~(size_t)255;
    return p;
  };
  float* stats = (float*)alloc(256 * 2 * sizeof(float));
  float* xres  = (float*)alloc((size_t)TOKENS * C_DIM * 4);
  unsigned short* abuf = (unsigned short*)alloc((size_t)TOKENS * C_DIM * 2);
  unsigned short* qbuf = (unsigned short*)alloc((size_t)TOKENS * C_DIM * 2);
  // big region (42 MB): k/v for self-attn + ctx k/v, later reused as geglu h
  unsigned short* big  = (unsigned short*)alloc((size_t)TOKENS * 2560 * 2);
  unsigned short* kbuf = big;
  unsigned short* vbuf = big + (size_t)TOKENS * C_DIM;
  unsigned short* kctx = big + (size_t)2 * TOKENS * C_DIM;
  unsigned short* vctx = kctx + (size_t)616 * C_DIM;
  unsigned short* hbuf = big;   // aliases k/v after they are dead

  // weights: convert + transpose to [N][K] bf16 (once per launch)
  auto cvtT = [&](const float* src, int K_, int N_) -> unsigned short* {
    unsigned short* dst = (unsigned short*)alloc((size_t)K_ * N_ * 2);
    cvt_transpose_kernel<<<dim3(N_ / 32, K_ / 32), dim3(32, 8), 0, stream>>>(
        src, dst, K_, N_);
    return dst;
  };
  auto cvt = [&](const float* src, size_t n) -> unsigned short* {
    unsigned short* dst = (unsigned short*)alloc(n * 2);
    f32_to_bf16_kernel<<<(int)((n + 255) / 256), 256, 0, stream>>>(src, dst, (int)n);
    return dst;
  };
  unsigned short* bW1 = cvtT(w_proj1, 640, 640);
  unsigned short* bQ1 = cvtT(wq1, 640, 640);
  unsigned short* bK1 = cvtT(wk1, 640, 640);
  unsigned short* bV1 = cvtT(wv1, 640, 640);
  unsigned short* bO1 = cvtT(wo1, 640, 640);
  unsigned short* bQ2 = cvtT(wq2, 640, 640);
  unsigned short* bK2 = cvtT(wk2, 768, 640);
  unsigned short* bV2 = cvtT(wv2, 768, 640);
  unsigned short* bO2 = cvtT(wo2, 640, 640);
  unsigned short* bGe = cvtT(w_geglu, 640, 5120);
  unsigned short* bFf = cvtT(w_ffout, 2560, 640);
  unsigned short* bW2 = cvtT(w_proj2, 640, 640);
  unsigned short* bCx = cvt(context, (size_t)616 * 768);   // A operand: row-major

  auto gemm = [&](const unsigned short* A, int lda, const unsigned short* Bt,
                  const float* bias, const float* res, float* oF, unsigned short* oB,
                  int ldo, int M, int N, int K) {
    dim3 g(N / 64, (M + 127) / 128);
    gemm_kernel<<<g, 256, 0, stream>>>(A, lda, Bt, bias, res, oF, oB, ldo, M, N, K);
  };
  const float kScale = 0.11180339887498949f;  // 1/sqrt(80)

  // 1) GroupNorm -> proj1 (residual stream, f32)
  gn_stats_kernel<<<256, 256, 0, stream>>>(inputs, stats);
  gn_apply_kernel<<<(TOKENS * C_DIM) / 256, 256, 0, stream>>>(inputs, stats, gn_g, gn_b, abuf);
  gemm(abuf, 640, bW1, b_proj1, nullptr, xres, nullptr, 640, TOKENS, 640, 640);

  // 2) self-attention
  layernorm_kernel<<<TOKENS / 8, 256, 0, stream>>>(xres, ln1_g, ln1_b, abuf, TOKENS);
  gemm(abuf, 640, bQ1, nullptr, nullptr, nullptr, qbuf, 640, TOKENS, 640, 640);
  gemm(abuf, 640, bK1, nullptr, nullptr, nullptr, kbuf, 640, TOKENS, 640, 640);
  gemm(abuf, 640, bV1, nullptr, nullptr, nullptr, vbuf, 640, TOKENS, 640, 640);
  attn_kernel<<<dim3(8, 8, 8), 256, 0, stream>>>(qbuf, kbuf, vbuf, abuf,
                                                 1024, 1024 * 640, 1024 * 640, kScale);
  gemm(abuf, 640, bO1, bo1, xres, xres, nullptr, 640, TOKENS, 640, 640);

  // 3) cross-attention (context: 8 x 77 x 768)
  layernorm_kernel<<<TOKENS / 8, 256, 0, stream>>>(xres, ln2_g, ln2_b, abuf, TOKENS);
  gemm(abuf, 640, bQ2, nullptr, nullptr, nullptr, qbuf, 640, TOKENS, 640, 640);
  gemm(bCx, 768, bK2, nullptr, nullptr, nullptr, kctx, 640, 616, 640, 768);
  gemm(bCx, 768, bV2, nullptr, nullptr, nullptr, vctx, 640, 616, 640, 768);
  attn_kernel<<<dim3(8, 8, 8), 256, 0, stream>>>(qbuf, kctx, vctx, abuf,
                                                 77, 1024 * 640, 77 * 640, kScale);
  gemm(abuf, 640, bO2, bo2, xres, xres, nullptr, 640, TOKENS, 640, 640);

  // 4) GEGLU FFN (fused gate epilogue, h in bf16 over reused big region)
  layernorm_kernel<<<TOKENS / 8, 256, 0, stream>>>(xres, ln3_g, ln3_b, abuf, TOKENS);
  gemm_geglu_kernel<<<dim3(2560 / 64, TOKENS / 128), 256, 0, stream>>>(
      abuf, 640, bGe, b_geglu, hbuf, TOKENS, 2560, 640);
  gemm(hbuf, 2560, bFf, b_ffout, xres, xres, nullptr, 640, TOKENS, 640, 2560);

  // 5) output projection + input residual -> d_out (f32)
  f32_to_bf16_kernel<<<(TOKENS * C_DIM) / 256, 256, 0, stream>>>(xres, abuf, TOKENS * C_DIM);
  gemm(abuf, 640, bW2, b_proj2, inputs, (float*)d_out, nullptr, 640, TOKENS, 640, 640);
}
